// RpnTarget_49400713839439
// MI455X (gfx1250) — compile-verified, matrix operand order checked by reading, not run on hardware
//
#include <hip/hip_runtime.h>
#include <stdint.h>

#define BATCH 4
#define GMAX 64
#define NA 262144
#define TT 256
#define PP 128
#define CHUNK 4096
#define K1THREADS 512

typedef __attribute__((ext_vector_type(16))) _Float16 v16h;
typedef __attribute__((ext_vector_type(8)))  float    v8f;

struct TF2 { unsigned a, b; };

__host__ __device__ constexpr unsigned rotl32(unsigned x, int r) {
    return (x << r) | (x >> (32 - r));
}

// JAX threefry2x32 (20 rounds), matches jax._src.prng
__host__ __device__ constexpr TF2 threefry2x32(unsigned k0, unsigned k1,
                                               unsigned x0, unsigned x1) {
    unsigned ks0 = k0, ks1 = k1, ks2 = 0x1BD11BDAu ^ k0 ^ k1;
    const int ra[4] = {13, 15, 26, 6};
    const int rb[4] = {17, 29, 16, 24};
    x0 += ks0; x1 += ks1;
    for (int i = 0; i < 4; ++i) { x0 += x1; x1 = rotl32(x1, ra[i]); x1 ^= x0; }
    x0 += ks1; x1 += ks2 + 1u;
    for (int i = 0; i < 4; ++i) { x0 += x1; x1 = rotl32(x1, rb[i]); x1 ^= x0; }
    x0 += ks2; x1 += ks0 + 2u;
    for (int i = 0; i < 4; ++i) { x0 += x1; x1 = rotl32(x1, ra[i]); x1 ^= x0; }
    x0 += ks0; x1 += ks1 + 3u;
    for (int i = 0; i < 4; ++i) { x0 += x1; x1 = rotl32(x1, rb[i]); x1 ^= x0; }
    x0 += ks1; x1 += ks2 + 4u;
    for (int i = 0; i < 4; ++i) { x0 += x1; x1 = rotl32(x1, ra[i]); x1 ^= x0; }
    x0 += ks2; x1 += ks0 + 5u;
    return TF2{x0, x1};
}

// key(42) -> (0,42); fold_in(k,0)/fold_in(k,1) -> threefry(k, seed(data))
constexpr TF2 KPOS = threefry2x32(0u, 42u, 0u, 0u);
constexpr TF2 KNEG = threefry2x32(0u, 42u, 0u, 1u);

// jax.random.uniform over flat (B*NA): counts split in halves, pair (i, i+N/2)
__device__ __forceinline__ float jax_uniform(TF2 key, unsigned i) {
    const unsigned HALF = (BATCH * NA) / 2u;
    unsigned c0, c1; bool hi;
    if (i < HALF) { c0 = i; c1 = i + HALF; hi = false; }
    else          { c0 = i - HALF; c1 = i; hi = true; }
    TF2 o = threefry2x32(key.a, key.b, c0, c1);
    unsigned bits = hi ? o.b : o.a;
    unsigned f = (bits >> 9) | 0x3F800000u;
    return __uint_as_float(f) - 1.0f;
}

// Sort key: descending priority, ascending anchor index on ties; masked -> 0
__device__ __forceinline__ unsigned long long make_key(bool m, float pri, unsigned idx) {
    unsigned ord = m ? (__float_as_uint(pri) + 1u) : 0u;
    return ((unsigned long long)ord << 32) | (unsigned)(0xFFFFFFFFu - idx);
}

// gfx1250 async global->LDS staging (ASYNCcnt-tracked)
__device__ __forceinline__ void async_copy_b32(const float* g, const float* lds) {
    unsigned l = (unsigned)(unsigned long long)(uintptr_t)lds;   // low 32 bits = LDS offset
    unsigned long long ga = (unsigned long long)(uintptr_t)g;
    asm volatile("global_load_async_to_lds_b32 %0, %1, off"
                 :: "v"(l), "v"(ga) : "memory");
}
__device__ __forceinline__ void wait_async0() {
    asm volatile("s_wait_asynccnt 0x0" ::: "memory");
}

// In-LDS ascending bitonic sort of CHUNK u64 keys
__device__ void bitonic_sort_asc(unsigned long long* s) {
    const int tid = threadIdx.x;
    for (int k = 2; k <= CHUNK; k <<= 1) {
        for (int j = k >> 1; j > 0; j >>= 1) {
            __syncthreads();
            for (int i = tid; i < CHUNK; i += blockDim.x) {
                int ixj = i ^ j;
                if (ixj > i) {
                    unsigned long long a = s[i], b = s[ixj];
                    bool asc = ((i & k) == 0);
                    if ((a > b) == asc) { s[i] = b; s[ixj] = a; }
                }
            }
        }
    }
    __syncthreads();
}

// Kernel 0: zero the pos/neg counters
__global__ void k0_init(int* counts) {
    if (threadIdx.x < BATCH * 2) counts[threadIdx.x] = 0;
}

// Kernel 1: IoU max/argmax + priority keys + block-level top-256 (pos & neg)
__global__ __launch_bounds__(K1THREADS) void k1_iou_keys(
    const float* __restrict__ gt_boxes, const float* __restrict__ gt_cls,
    const float* __restrict__ anchors, uint8_t* __restrict__ argmax_out,
    unsigned long long* __restrict__ posTop, unsigned long long* __restrict__ negTop,
    int* __restrict__ counts)
{
    __shared__ float sGT[GMAX * 5];
    __shared__ float sTag[GMAX];
    __shared__ float sIou[CHUNK];
    __shared__ unsigned long long sKey[CHUNK];
    __shared__ int sPos, sNeg;

    const int b = blockIdx.y;
    const int blk = blockIdx.x;          // 0..63
    const int t = threadIdx.x;
    if (t == 0) { sPos = 0; sNeg = 0; }

    if (t < GMAX * 5) async_copy_b32(gt_boxes + b * GMAX * 5 + t, &sGT[t]);
    if (t < GMAX)     async_copy_b32(gt_cls + b * GMAX * 3 + t * 3 + 2, &sTag[t]);
    wait_async0();
    __syncthreads();

    int pcnt = 0, ncnt = 0;
    const float4* anc = reinterpret_cast<const float4*>(anchors);
    for (int q = 0; q < CHUNK / K1THREADS; ++q) {
        const int la = q * K1THREADS + t;
        const int a  = blk * CHUNK + la;
        const float4 av = anc[b * NA + a];
        const float area_an = (av.w - av.y) * (av.z - av.x);
        float best = -__builtin_inff();
        int barg = 0;
        for (int g = 0; g < GMAX; ++g) {
            float g0 = sGT[g * 5 + 0], g1 = sGT[g * 5 + 1];
            float g2 = sGT[g * 5 + 2], g3 = sGT[g * 5 + 3];
            float iw = fmaxf(0.0f, fminf(g3, av.w) - fmaxf(g1, av.y));
            float ih = fmaxf(0.0f, fminf(g2, av.z) - fmaxf(g0, av.x));
            float inter = iw * ih;
            float area_gt = (g3 - g1) * (g2 - g0);
            float iou = inter / (area_gt + area_an - inter);
            float v = (sTag[g] != -1.0f) ? iou : -1.0f;
            if (v > best) { best = v; barg = g; }
        }
        argmax_out[b * NA + a] = (uint8_t)barg;
        const bool pos = best > 0.6f;
        const bool neg = best < 0.3f;
        pcnt += pos ? 1 : 0;
        ncnt += neg ? 1 : 0;
        sIou[la] = best;
        float pp = jax_uniform(KPOS, (unsigned)(b * NA + a));
        sKey[la] = make_key(pos, pp, (unsigned)a);
    }
    atomicAdd(&sPos, pcnt);
    atomicAdd(&sNeg, ncnt);

    bitonic_sort_asc(sKey);
    if (t < 256) posTop[(b * 64 + blk) * 256 + t] = sKey[CHUNK - 1 - t];
    __syncthreads();

    for (int q = 0; q < CHUNK / K1THREADS; ++q) {
        const int la = q * K1THREADS + t;
        const int a  = blk * CHUNK + la;
        const bool neg = sIou[la] < 0.3f;
        float pn = jax_uniform(KNEG, (unsigned)(b * NA + a));
        sKey[la] = make_key(neg, pn, (unsigned)a);
    }
    bitonic_sort_asc(sKey);
    if (t < 256) negTop[(b * 64 + blk) * 256 + t] = sKey[CHUNK - 1 - t];
    if (t == 0) {
        atomicAdd(&counts[b * 2 + 0], sPos);
        atomicAdd(&counts[b * 2 + 1], sNeg);
    }
}

// Kernel 2: merge round — top-256 of each 4096-chunk (jobs: 0..3 pos, 4..7 neg)
__global__ __launch_bounds__(K1THREADS) void k_select(
    const unsigned long long* __restrict__ inPos, const unsigned long long* __restrict__ inNeg,
    unsigned long long* __restrict__ outPos, unsigned long long* __restrict__ outNeg,
    int inPerJob)
{
    __shared__ unsigned long long sKey[CHUNK];
    const int job = blockIdx.y;
    const int b = job & 3;
    const bool isNeg = job >= 4;
    const unsigned long long* in = isNeg ? inNeg : inPos;
    unsigned long long* out = isNeg ? outNeg : outPos;
    const int blk = blockIdx.x;
    const int start = blk * CHUNK;
    for (int i = threadIdx.x; i < CHUNK; i += blockDim.x) {
        int gi = start + i;
        sKey[i] = (gi < inPerJob) ? in[b * inPerJob + gi] : 0ull;
    }
    bitonic_sort_asc(sKey);
    const int outBlocks = gridDim.x;
    if (threadIdx.x < 256)
        out[(b * outBlocks + blk) * 256 + threadIdx.x] = sKey[CHUNK - 1 - threadIdx.x];
}

// Kernel 3: final assembly; class gather via v_wmma_f32_16x16x32_f16 (exact 0/1 data)
__global__ __launch_bounds__(TT) void k3_assemble(
    const float* __restrict__ gt_boxes, const float* __restrict__ gt_cls,
    const float* __restrict__ anchors, const uint8_t* __restrict__ argmax_in,
    const unsigned long long* __restrict__ posFin, const unsigned long long* __restrict__ negFin,
    const int* __restrict__ counts, float* __restrict__ out)
{
    __shared__ float sGT[GMAX * 5];
    __shared__ float sCls[GMAX * 3];
    __shared__ int   sGidx[TT];
    __shared__ float sHits[GMAX];
    // B fragment pre-permuted into WMMA operand order: [ck][hi][n][s], f16
    __shared__ __align__(32) _Float16 sBfrag[2 * 2 * 16 * 16];

    const int b = blockIdx.x;
    const int j = threadIdx.x;

    if (j < GMAX * 5) async_copy_b32(gt_boxes + b * GMAX * 5 + j, &sGT[j]);
    if (j < GMAX * 3) async_copy_b32(gt_cls + b * GMAX * 3 + j, &sCls[j]);
    wait_async0();
    if (j < GMAX) sHits[j] = 0.0f;
    __syncthreads();

    // Build B fragment (1024 f16): sBfrag[((ck*2+hi)*16+n)*16+s] = n<2 ? cls[kg][n] : 0
#pragma unroll
    for (int e = 0; e < 4; ++e) {
        int idx = e * TT + j;
        int s  = idx & 15;
        int n_ = (idx >> 4) & 15;
        int hi_ = (idx >> 8) & 1;
        int ck_ = idx >> 9;
        int kl = ((s >> 3) << 4) + (hi_ << 3) + (s & 7);
        int kg = ck_ * 32 + kl;
        float bv = (n_ < 2) ? sCls[kg * 3 + n_] : 0.0f;
        sBfrag[idx] = (_Float16)bv;
    }

    const int posTotal = counts[b * 2 + 0];
    const int negTotal = counts[b * 2 + 1];
    const int pos_count = min(PP, posTotal);
    const int neg_count = min(TT - pos_count, negTotal);
    const bool is_pos = j < pos_count;
    const bool is_neg = (j >= pos_count) && (j < pos_count + neg_count);

    const int jp = min(j, PP - 1);
    const int jn = min(max(j - pos_count, 0), TT - 1);
    const unsigned long long key = is_pos ? posFin[b * TT + jp] : negFin[b * TT + jn];
    const int anchor_idx = (int)(0xFFFFFFFFu - (unsigned)(key & 0xFFFFFFFFull));
    const int gt = is_pos ? (int)argmax_in[b * NA + anchor_idx] : 0;

    float d0 = 0.f, d1 = 0.f, d2 = 0.f, d3 = 0.f;
    if (is_pos) {
        const float4 av = reinterpret_cast<const float4*>(anchors)[b * NA + anchor_idx];
        float g0 = sGT[gt * 5 + 0], g1 = sGT[gt * 5 + 1];
        float g2 = sGT[gt * 5 + 2], g3 = sGT[gt * 5 + 3];
        float h = av.z - av.x, w = av.w - av.y;
        float gh = g2 - g0, gw = g3 - g1;
        float cy = (av.z + av.x) * 0.5f, cx = (av.w + av.y) * 0.5f;
        float gcy = (g2 + g0) * 0.5f,   gcx = (g3 + g1) * 0.5f;
        d0 = ((gcy - cy) / h) / 0.1f;
        d1 = ((gcx - cx) / w) / 0.1f;
        d2 = logf(gh / h) / 0.2f;
        d3 = logf(gw / w) / 0.2f;
        sHits[gt] = 1.0f;
    }

    float* outClass = out;
    float* outDelta = out + BATCH * TT * 2;
    float* outIdx   = out + BATCH * TT * 2 + BATCH * TT * 4;
    float* outGtNum = outIdx + BATCH * TT * 2;
    float* outPosC  = outGtNum + BATCH;
    float* outMiss  = outPosC + BATCH;

    const int r = b * TT + j;
    outDelta[r * 4 + 0] = d0; outDelta[r * 4 + 1] = d1;
    outDelta[r * 4 + 2] = d2; outDelta[r * 4 + 3] = d3;
    const int out_idx = (is_pos || is_neg) ? anchor_idx : -1;
    const int tag = is_pos ? 1 : (is_neg ? 0 : -1);
    outIdx[r * 2 + 0] = (float)out_idx;
    outIdx[r * 2 + 1] = (float)tag;

    sGidx[j] = is_pos ? gt : 127;  // sentinel never matches k in [0,64)
    __syncthreads();

    // class_ids[256x2] = onehot(gt_idx)[256x64] x cls[64x2], exact in f16/f32
    const int wave = j >> 5;
    const int lane = j & 31;
    const int hi = lane >> 4;
    const int n = lane & 15;

    // B fragments: depend only on (ck, hi, n) -> hoisted, one 32B vector load each
    const v16h bm0 = *reinterpret_cast<const v16h*>(&sBfrag[(hi * 16 + n) * 16]);
    const v16h bm1 = *reinterpret_cast<const v16h*>(&sBfrag[((2 + hi) * 16 + n) * 16]);

#pragma unroll
    for (int tt = 0; tt < 2; ++tt) {
        const int tile = wave * 2 + tt;
        const int gid = sGidx[tile * 16 + n];   // A row id for this lane
        v16h am0, am1;
#pragma unroll
        for (int s = 0; s < 16; ++s) {
            int kl = ((s >> 3) << 4) + (hi << 3) + (s & 7);
            am0[s] = (_Float16)((gid == kl)      ? 1.0f : 0.0f);
            am1[s] = (_Float16)((gid == 32 + kl) ? 1.0f : 0.0f);
        }
        v8f acc = {0.f, 0.f, 0.f, 0.f, 0.f, 0.f, 0.f, 0.f};
        acc = __builtin_amdgcn_wmma_f32_16x16x32_f16(
            false, am0, false, bm0, (short)0, acc, false, false);
        acc = __builtin_amdgcn_wmma_f32_16x16x32_f16(
            false, am1, false, bm1, (short)0, acc, false, false);
        if (n < 2) {
#pragma unroll
            for (int rr = 0; rr < 8; ++rr) {
                int row = tile * 16 + hi * 8 + rr;
                bool rneg = (row >= pos_count) && (row < pos_count + neg_count);
                float v = acc[rr] + ((rneg && n == 0) ? 1.0f : 0.0f);
                outClass[(b * TT + row) * 2 + n] = v;
            }
        }
    }
    __syncthreads();

    if (j == 0) {
        float gt_num = 0.f, hitsum = 0.f;
        for (int g = 0; g < GMAX; ++g) {
            gt_num += (sCls[g * 3 + 2] != -1.0f) ? 1.0f : 0.0f;
            hitsum += sHits[g];
        }
        outGtNum[b] = gt_num;
        outPosC[b] = (float)pos_count;
        outMiss[b] = gt_num - hitsum;
    }
}

extern "C" void kernel_launch(void* const* d_in, const int* in_sizes, int n_in,
                              void* d_out, int out_size, void* d_ws, size_t ws_size,
                              hipStream_t stream) {
    (void)in_sizes; (void)n_in; (void)out_size; (void)ws_size;
    const float* gt_boxes = (const float*)d_in[0];
    const float* gt_cls   = (const float*)d_in[1];
    const float* anchors  = (const float*)d_in[2];

    unsigned long long* posTop1 = (unsigned long long*)d_ws;        // 4*64*256
    unsigned long long* negTop1 = posTop1 + BATCH * 64 * 256;
    unsigned long long* posTop2 = negTop1 + BATCH * 64 * 256;       // 4*4*256
    unsigned long long* negTop2 = posTop2 + BATCH * 4 * 256;
    unsigned long long* posFin  = negTop2 + BATCH * 4 * 256;        // 4*256
    unsigned long long* negFin  = posFin + BATCH * TT;
    int*      counts = (int*)(negFin + BATCH * TT);                 // 8 ints (+pad)
    uint8_t*  argmax = (uint8_t*)(counts + 16);                     // B*NA bytes

    k0_init<<<1, 32, 0, stream>>>(counts);
    k1_iou_keys<<<dim3(NA / CHUNK, BATCH), K1THREADS, 0, stream>>>(
        gt_boxes, gt_cls, anchors, argmax, posTop1, negTop1, counts);
    k_select<<<dim3(4, 8), K1THREADS, 0, stream>>>(posTop1, negTop1, posTop2, negTop2, 64 * 256);
    k_select<<<dim3(1, 8), K1THREADS, 0, stream>>>(posTop2, negTop2, posFin, negFin, 4 * 256);
    k3_assemble<<<BATCH, TT, 0, stream>>>(
        gt_boxes, gt_cls, anchors, argmax, posFin, negFin, counts, (float*)d_out);
}
